// SimpleMoELayer_54219667144995
// MI455X (gfx1250) — matrix operand order repaired and checked by reading, not compiled, not run on hardware
//
#include <hip/hip_runtime.h>

// Problem constants (from reference): B=2, S=2048, D=1024, F=4096, E=8, top_k=2
#define D_DIM 1024
#define F_DIM 4096
#define E_NUM 8
#define MT    32            // tokens per tile (2 x WMMA M) -> halves L2 weight traffic
#define FC    128           // F-chunk per step (8 N-tiles x 16)
#define LDX   (D_DIM + 8)   // padded LDS stride (elements), keeps 16B alignment
#define LDH   (FC + 8)
#define NTHREADS 512        // 16 waves

typedef __attribute__((ext_vector_type(16))) __bf16 v16bf;
typedef __attribute__((ext_vector_type(8)))  __bf16 v8bf;
typedef __attribute__((ext_vector_type(8)))  float  v8f;

__device__ inline __bf16 f2bf(float f) {
    unsigned u = __builtin_bit_cast(unsigned, f);
    u += 0x7FFFu + ((u >> 16) & 1u);          // round-to-nearest-even
    unsigned short h = (unsigned short)(u >> 16);
    return __builtin_bit_cast(__bf16, h);
}

// A-fragment (16x32 bf16) from LDS, per CDNA5 ISA 16-bit A layout:
// lane<16: row M=lane, K = {k0+0..7, k0+16..23}; lane>=16: K = {k0+8..15, k0+24..31}
__device__ inline v16bf load_a(const __bf16* base, int ld, int k0, int lane) {
    int r  = lane & 15;
    int kq = (lane >> 4) << 3;                // 0 or 8
    const __bf16* p = base + r * ld + k0 + kq;
    v8bf lo = *(const v8bf*)(p);              // ds_load_b128
    v8bf hi = *(const v8bf*)(p + 16);         // ds_load_b128
    v16bf a;
#pragma unroll
    for (int i = 0; i < 8; ++i) { a[i] = lo[i]; a[8 + i] = hi[i]; }
    return a;
}

// B-fragment (32x16) from pre-transposed bf16 weights Wt[N][K] (ldk = K-dim).
// lane<16: col N = n0+lane, K = {k0+0..7, k0+16..23} ; lane>=16: col N = n0+lane-16, K = {k0+8..15, k0+24..31}.
// Per lane: two runs of 8 contiguous bf16 (at kb and kb+16) = two global_load_b128, no conversion VALU.
__device__ inline v16bf load_bt(const __bf16* __restrict__ Wt, int ldk,
                                int k0, int n0, int lane) {
    int n  = n0 + (lane & 15);
    int kb = k0 + ((lane >> 4) << 3);         // 0 or 8
    const __bf16* p = Wt + (size_t)n * ldk + kb;
    v8bf lo = *(const v8bf*)(p);
    v8bf hi = *(const v8bf*)(p + 16);
    v16bf b;
#pragma unroll
    for (int i = 0; i < 8; ++i) { b[i] = lo[i]; b[8 + i] = hi[i]; }
    return b;
}

__global__ void moe_init(float* __restrict__ out, int nOut, int* __restrict__ counts) {
    int i = blockIdx.x * blockDim.x + threadIdx.x;
    if (i < nOut) out[i] = 0.0f;
    if (i < E_NUM) counts[i] = 0;
}

// Convert f32 [R][C] (per expert) -> bf16 [C][R], coalesced both sides via LDS tile.
__global__ __launch_bounds__(256) void cvt_transpose(
    const float* __restrict__ src, __bf16* __restrict__ dst, int R, int C)
{
    __shared__ __bf16 tile[64][64 + 4];
    const size_t ebase = (size_t)blockIdx.z * R * C;
    const int r0 = blockIdx.y * 64, c0 = blockIdx.x * 64;
    const float* s = src + ebase + (size_t)r0 * C + c0;
#pragma unroll
    for (int i = threadIdx.x; i < 64 * 64; i += 256) {
        int tr = i >> 6, tc = i & 63;                    // coalesced along tc
        tile[tr][tc] = f2bf(s[(size_t)tr * C + tc]);
    }
    __syncthreads();
    __bf16* d = dst + ebase + (size_t)c0 * R + r0;
#pragma unroll
    for (int i = threadIdx.x; i < 64 * 64; i += 256) {
        int tc = i >> 6, tr = i & 63;                    // coalesced along tr
        d[(size_t)tc * R + tr] = tile[tr][tc];
    }
}

__global__ __launch_bounds__(256) void moe_router(
    const float* __restrict__ X, const float* __restrict__ Wr,
    int* __restrict__ counts, int* __restrict__ lists, int nTokens)
{
    int t = blockIdx.x * blockDim.x + threadIdx.x;
    if (t >= nTokens) return;
    const float* x = X + (size_t)t * D_DIM;
    float b1 = -3.4e38f, b2 = -3.4e38f;
    int e1 = 0, e2 = 0;
#pragma unroll 1
    for (int e = 0; e < E_NUM; ++e) {
        const float* w = Wr + (size_t)e * D_DIM;
        float s = 0.0f;
#pragma unroll 4
        for (int d = 0; d < D_DIM; d += 4) {
            s = __builtin_fmaf(x[d],     w[d],     s);
            s = __builtin_fmaf(x[d + 1], w[d + 1], s);
            s = __builtin_fmaf(x[d + 2], w[d + 2], s);
            s = __builtin_fmaf(x[d + 3], w[d + 3], s);
        }
        if (s > b1)      { b2 = b1; e2 = e1; b1 = s; e1 = e; }
        else if (s > b2) { b2 = s;  e2 = e; }
    }
    int slot = atomicAdd(&counts[e1], 1);
    lists[e1 * nTokens + slot] = t;
    slot = atomicAdd(&counts[e2], 1);
    lists[e2 * nTokens + slot] = t;
}

__global__ __launch_bounds__(NTHREADS) void moe_ffn(
    const float* __restrict__ X,
    const __bf16* __restrict__ W1t,   // [E][F][D] bf16 (transposed)
    const __bf16* __restrict__ W2t,   // [E][D][F] bf16 (transposed)
    const int* __restrict__ counts,
    const int* __restrict__ lists, float* __restrict__ out, int nTokens)
{
    __shared__ __bf16 Xs[MT * LDX];   // 32 x 1024 bf16 token tile  (~66 KB)
    __shared__ __bf16 Hs[MT * LDH];   // 32 x 128 bf16 SiLU chunk   (~8.7 KB)
    __shared__ int    toks[MT];
    __shared__ int    s_cnt;

    const int e    = blockIdx.y;
    const int tile = blockIdx.x;
    if (threadIdx.x == 0) s_cnt = counts[e];
    __syncthreads();
    const int cnt = s_cnt;
    const int t0  = tile * MT;
    if (t0 >= cnt) return;                       // uniform across block
    const int nvalid = (cnt - t0 < MT) ? (cnt - t0) : MT;

    if (threadIdx.x < MT)
        toks[threadIdx.x] = (threadIdx.x < nvalid)
                          ? lists[e * nTokens + t0 + threadIdx.x] : -1;
    __syncthreads();

    // Stage token tile into LDS as bf16, zero-pad invalid rows.
    for (int idx = threadIdx.x; idx < MT * D_DIM; idx += NTHREADS) {
        int m = idx >> 10;                       // / 1024
        int d = idx & (D_DIM - 1);
        int tok = toks[m];
        float v = (tok >= 0) ? X[(size_t)tok * D_DIM + d] : 0.0f;
        Xs[m * LDX + d] = f2bf(v);
    }
    __syncthreads();

    const int lane = threadIdx.x & 31;
    const int wave = threadIdx.x >> 5;           // 0..15
    const int mh   = wave & 1;                   // M-half for phase A
    const int nh   = wave >> 1;                  // N-tile (0..7) for phase A
    const __bf16* W1e = W1t + (size_t)e * D_DIM * F_DIM;   // [F][D]
    const __bf16* W2e = W2t + (size_t)e * F_DIM * D_DIM;   // [D][F]

    const v8f vzero = {0.f, 0.f, 0.f, 0.f, 0.f, 0.f, 0.f, 0.f};
    v8f acc[8];   // 4 N-tiles x 2 M-halves of persistent output accumulators
#pragma unroll
    for (int i = 0; i < 8; ++i) acc[i] = vzero;

    for (int fc = 0; fc < F_DIM; fc += FC) {
        // ---- Phase A: one 16x16 H tile per wave:
        //      rows [mh*16, +16), cols [fc + nh*16, +16)
        const int n0 = fc + nh * 16;
        const __bf16* Xh = Xs + (size_t)mh * 16 * LDX;
        v8f h = vzero;
#pragma unroll 4
        for (int k = 0; k < D_DIM; k += 32) {
            __builtin_prefetch(W1e + (size_t)(n0 + (lane & 15)) * D_DIM + k + 128, 0, 1);
            v16bf a = load_a(Xh, LDX, k, lane);
            v16bf b = load_bt(W1e, D_DIM, k, n0, lane);
            h = __builtin_amdgcn_wmma_f32_16x16x32_bf16(
                    false, a, false, b, (short)0, h, false, false);
        }
        // C layout: VGPR r -> M = r (lane<16) or r+8 (lane>=16); N = lane&15
#pragma unroll
        for (int r = 0; r < 8; ++r) {
            int M = mh * 16 + ((lane < 16) ? r : (r + 8));
            float x = h[r];
            float s = x * __builtin_amdgcn_rcpf(1.0f + __expf(-x));   // fast SiLU
            Hs[M * LDH + nh * 16 + (lane & 15)] = f2bf(s);
        }
        __syncthreads();

        // ---- Phase B: acc += H @ W2[fc:fc+FC, nd:nd+16]
        // Two A-fragments (M-halves) reused across 4 N-tiles; each B-fragment
        // feeds two WMMAs (shared by both M-halves) -> halved B traffic.
#pragma unroll
        for (int k = 0; k < FC; k += 32) {
            v16bf a0 = load_a(Hs,                        LDH, k, lane);
            v16bf a1 = load_a(Hs + (size_t)16 * LDH,     LDH, k, lane);
#pragma unroll
            for (int j = 0; j < 4; ++j) {
                const int nd = (wave + j * 16) * 16;     // 64 N-tiles over 16 waves
                v16bf b = load_bt(W2e, F_DIM, fc + k, nd, lane);
                acc[j * 2]     = __builtin_amdgcn_wmma_f32_16x16x32_bf16(
                                     false, a0, false, b, (short)0, acc[j * 2], false, false);
                acc[j * 2 + 1] = __builtin_amdgcn_wmma_f32_16x16x32_bf16(
                                     false, a1, false, b, (short)0, acc[j * 2 + 1], false, false);
            }
        }
        __syncthreads();                         // Hs reused next chunk
    }

    // Combine: unweighted sum over selected experts -> f32 global atomics
#pragma unroll
    for (int j = 0; j < 4; ++j) {
        const int nd = (wave + j * 16) * 16 + (lane & 15);
#pragma unroll
        for (int half = 0; half < 2; ++half) {
#pragma unroll
            for (int r = 0; r < 8; ++r) {
                int M = half * 16 + ((lane < 16) ? r : (r + 8));
                if (M < nvalid) {
                    int tok = toks[M];
                    atomicAdd(&out[(size_t)tok * D_DIM + nd], acc[j * 2 + half][r]);
                }
            }
        }
    }
}

extern "C" void kernel_launch(void* const* d_in, const int* in_sizes, int n_in,
                              void* d_out, int out_size, void* d_ws, size_t ws_size,
                              hipStream_t stream)
{
    const float* X  = (const float*)d_in[0];     // [B,S,D]
    const float* Wr = (const float*)d_in[1];     // [E,D]
    const float* W1 = (const float*)d_in[2];     // [E,D,F]
    const float* W2 = (const float*)d_in[3];     // [E,F,D]
    float* out = (float*)d_out;                  // [B,S,D]

    const int nTokens = in_sizes[0] / D_DIM;     // B*S = 4096

    // Workspace layout: counts | lists | W1t bf16 | W2t bf16
    char* ws = (char*)d_ws;
    int* counts = (int*)ws;                                   // 16 ints
    int* lists  = counts + 16;                                // E * nTokens ints
    size_t off = 64 + (size_t)E_NUM * nTokens * 4;
    off = (off + 255) & ~(size_t)255;
    __bf16* W1t = (__bf16*)(ws + off);                        // [E][F][D]
    off += (size_t)E_NUM * D_DIM * F_DIM * 2;
    __bf16* W2t = (__bf16*)(ws + off);                        // [E][D][F]

    const int nOut = out_size;                   // nTokens * D
    moe_init<<<(nOut + 255) / 256, 256, 0, stream>>>(out, nOut, counts);

    // One-time weight convert+transpose to bf16 (amortized across all tiles)
    {
        dim3 g1(F_DIM / 64, D_DIM / 64, E_NUM);  // W1 [D][F] -> W1t [F][D]
        cvt_transpose<<<g1, 256, 0, stream>>>(W1, W1t, D_DIM, F_DIM);
        dim3 g2(D_DIM / 64, F_DIM / 64, E_NUM);  // W2 [F][D] -> W2t [D][F]
        cvt_transpose<<<g2, 256, 0, stream>>>(W2, W2t, F_DIM, D_DIM);
    }

    moe_router<<<(nTokens + 255) / 256, 256, 0, stream>>>(X, Wr, counts, lists, nTokens);

    dim3 grid((nTokens + MT - 1) / MT, E_NUM);   // worst-case tiles; empty tiles exit
    moe_ffn<<<grid, NTHREADS, 0, stream>>>(X, W1t, W2t, counts, lists, out, nTokens);
}
